// UnifiedRingStarBlock_46179488367254
// MI455X (gfx1250) — matrix operand order repaired and checked by reading, not compiled, not run on hardware
//
#include <hip/hip_runtime.h>
#include <cstdint>
#include <cstddef>

// ---------------------------------------------------------------------------
// UnifiedRingStarBlock for MI455X (gfx1250, wave32, WMMA f32_16x16x32_f16)
// ---------------------------------------------------------------------------

typedef __attribute__((ext_vector_type(16))) _Float16 v16h;
typedef __attribute__((ext_vector_type(8)))  float    v8f;

constexpr int   Bb   = 16;
constexpr int   Ls   = 2048;
constexpr int   Nc   = 512;
constexpr int   Dd   = 64;
constexpr int   Hh   = 16;
constexpr int   Cc   = 16;
constexpr int   BL   = Bb * Ls;            // 32768 rows
constexpr float BN_S = 0.999995000037499638f; // 1/sqrt(1+1e-5)
constexpr int   TOPK  = 27;                // max(16, log2(512)*3)
constexpr int   STOPK = 45;                // max(32, sqrt(512)*2)

#define GF_ACCUM 1
#define GF_BIAS  2
#define GF_GELU  4
#define GF_SIGM  8

__device__ __forceinline__ float gelu_f(float x) {
  return 0.5f * x * (1.0f + erff(x * 0.70710678118654752440f));
}
__device__ __forceinline__ float sigm_f(float x) { return 1.0f / (1.0f + expf(-x)); }

// ---------------------------------------------------------------------------
// Fragment loaders (contiguous k order used by this file, same for A and B).
//  A (16x32 f16): lane half h, elem e -> k = 16*h + e       (16 contiguous)
//  B (32x16 f16): lane half h, elem e -> k = 16*h + e       (16 contiguous)
// asel multiplies A by 0/1 for branch-free zero padding (folds away when 1.0f).
// ---------------------------------------------------------------------------
__device__ __forceinline__ v16h load_frag_a(const float* __restrict__ arow,
                                            int half, float asel) {
  alignas(16) float abuf[16];
  const float4* p0 = (const float4*)(arow + half * 16);
  const float4* p1 = (const float4*)(arow + half * 16 + 8);
  *(float4*)(abuf + 0)  = p0[0];
  *(float4*)(abuf + 4)  = p0[1];
  *(float4*)(abuf + 8)  = p1[0];
  *(float4*)(abuf + 12) = p1[1];
  v16h af;
#pragma unroll
  for (int e = 0; e < 16; ++e) af[e] = (_Float16)(abuf[e] * asel);
  return af;
}

__device__ __forceinline__ v16h load_frag_b(const float* __restrict__ wrow, int half) {
  alignas(16) float bbuf[16];
  const float4* q = (const float4*)(wrow + half * 16);
  *(float4*)(bbuf + 0)  = q[0];
  *(float4*)(bbuf + 4)  = q[1];
  *(float4*)(bbuf + 8)  = q[2];
  *(float4*)(bbuf + 12) = q[3];
  v16h bf;
#pragma unroll
  for (int e = 0; e < 16; ++e) bf[e] = (_Float16)bbuf[e];
  return bf;
}

__device__ __forceinline__ void gemm_epilogue(v8f acc, const float* __restrict__ bias,
                                              float* __restrict__ Y, int mBase, int half,
                                              int ocol, int Nout, float alpha, int flags) {
  float bv = (flags & GF_BIAS) ? bias[ocol] : 0.0f;
#pragma unroll
  for (int rr = 0; rr < 8; ++rr) {
    int m = mBase + rr + half * 8;     // D fragment: row m = rr + 8*half
    float v = (acc[rr] + bv) * alpha;
    size_t idx = (size_t)m * Nout + ocol;
    if (flags & GF_ACCUM) v += Y[idx];
    if (flags & GF_GELU)  v = gelu_f(v);
    if (flags & GF_SIGM)  v = sigm_f(v);
    Y[idx] = v;
  }
}

// ---------------------------------------------------------------------------
// Plain GEMM:  Y[m, o] = epi( sum_k A[m,k] * W[o*ldw + colOff + k] )
// Block: 256 threads = 8 waves; block tile 16(M) x 128(N); wave tile 16x16.
// ---------------------------------------------------------------------------
__global__ __launch_bounds__(256) void gemm_lin_kernel(
    const float* __restrict__ A, const float* __restrict__ W,
    const float* __restrict__ bias, float* __restrict__ Y,
    int K, int Nout, int ldw, int colOff, float alpha, int flags) {
  int wid  = threadIdx.x >> 5;
  int lane = threadIdx.x & 31;
  int mBase = blockIdx.y << 4;
  int nBase = ((int)blockIdx.x * 8 + wid) << 4;
  if (nBase >= Nout) return;
  int half = lane >> 4;
  int ocol = nBase + (lane & 15);

  const float* arow = A + (size_t)(mBase + (lane & 15)) * K;
  const float* wrow = W + (size_t)ocol * ldw + colOff;

  v8f acc = {};
  for (int k0 = 0; k0 < K; k0 += 64) {   // 2 WMMA per iteration (K % 64 == 0)
    v16h a0 = load_frag_a(arow + k0, half, 1.0f);
    v16h b0 = load_frag_b(wrow + k0, half);
    acc = __builtin_amdgcn_wmma_f32_16x16x32_f16(false, a0, false, b0,
                                                 (short)0, acc, false, false);
    v16h a1 = load_frag_a(arow + k0 + 32, half, 1.0f);
    v16h b1 = load_frag_b(wrow + k0 + 32, half);
    acc = __builtin_amdgcn_wmma_f32_16x16x32_f16(false, a1, false, b1,
                                                 (short)0, acc, false, false);
  }
  gemm_epilogue(acc, bias, Y, mBase, half, ocol, Nout, alpha, flags);
}

// ---------------------------------------------------------------------------
// Conv GEMM (full conv as tap-summed GEMM):
//   Y[(b,l), o] = sum_t sum_i Wt[(t*Nout+o)*K + i] * A[(b, l + dil*t - pad), i]
// Tap loop hoisted: shift/validity/base pointers computed once per tap;
// zero padding is branch-free via asel.
// ---------------------------------------------------------------------------
__global__ __launch_bounds__(256) void gemm_conv_kernel(
    const float* __restrict__ A, const float* __restrict__ W,
    float* __restrict__ Y, int K, int Nout,
    int taps, int dil, int pad, int Lseq) {
  int wid  = threadIdx.x >> 5;
  int lane = threadIdx.x & 31;
  int mBase = blockIdx.y << 4;
  int nBase = ((int)blockIdx.x * 8 + wid) << 4;
  if (nBase >= Nout) return;
  int half = lane >> 4;
  int r    = mBase + (lane & 15);
  int bb   = r / Lseq;
  int ll   = r - bb * Lseq;
  int ocol = nBase + (lane & 15);

  v8f acc = {};
  for (int tp = 0; tp < taps; ++tp) {
    int ls = ll + dil * tp - pad;
    bool valid = (ls >= 0) && (ls < Lseq);
    int lsc = valid ? ls : 0;                       // clamped, always in-bounds
    float asel = valid ? 1.0f : 0.0f;
    const float* arow = A + ((size_t)(bb * Lseq + lsc)) * K;
    const float* wrow = W + ((size_t)tp * Nout + ocol) * K;
    for (int i0 = 0; i0 < K; i0 += 64) {
      v16h a0 = load_frag_a(arow + i0, half, asel);
      v16h b0 = load_frag_b(wrow + i0, half);
      acc = __builtin_amdgcn_wmma_f32_16x16x32_f16(false, a0, false, b0,
                                                   (short)0, acc, false, false);
      v16h a1 = load_frag_a(arow + i0 + 32, half, asel);
      v16h b1 = load_frag_b(wrow + i0 + 32, half);
      acc = __builtin_amdgcn_wmma_f32_16x16x32_f16(false, a1, false, b1,
                                                   (short)0, acc, false, false);
    }
  }
  gemm_epilogue(acc, nullptr, Y, mBase, half, ocol, Nout, 1.0f, 0);
}

// ---------------------------------------------------------------------------
// Weight repack for full convs: W(O,K,T) -> Wt[(t*O+o)*K+i]
// ---------------------------------------------------------------------------
__global__ void repack_kernel(const float* __restrict__ Wsrc, float* __restrict__ Wt,
                              int O, int K, int T) {
  int idx = blockIdx.x * blockDim.x + threadIdx.x;
  int total = O * K * T;
  if (idx >= total) return;
  int t = idx % T;
  int i = (idx / T) % K;
  int o = idx / (T * K);
  Wt[((size_t)t * O + o) * K + i] = Wsrc[idx];
}

// ---------------------------------------------------------------------------
// Elementwise / small convs on (B,L,N) layout
// ---------------------------------------------------------------------------
__global__ void avgpool7_kernel(const float* __restrict__ X, float* __restrict__ Y) {
  int idx = blockIdx.x * blockDim.x + threadIdx.x;
  if (idx >= BL * Nc) return;
  int n = idx & (Nc - 1);
  int l = (idx >> 9) & (Ls - 1);
  int b = idx >> 20;
  float acc = 0.0f;
#pragma unroll
  for (int t = -3; t <= 3; ++t) {
    int lt = l + t;
    if (lt >= 0 && lt < Ls) acc += X[((size_t)(b * Ls + lt)) * Nc + n];
  }
  Y[idx] = acc * (1.0f / 7.0f);
}

__global__ void dwconv3_kernel(const float* __restrict__ X, const float* __restrict__ w,
                               const float* __restrict__ bsrc, float* __restrict__ Y, int act) {
  int idx = blockIdx.x * blockDim.x + threadIdx.x;
  if (idx >= BL * Nc) return;
  int n = idx & (Nc - 1);
  int l = (idx >> 9) & (Ls - 1);
  int b = idx >> 20;
  float acc = bsrc[n];
#pragma unroll
  for (int t = 0; t < 3; ++t) {
    int lt = l + t - 1;
    if (lt >= 0 && lt < Ls) acc += w[n * 3 + t] * X[((size_t)(b * Ls + lt)) * Nc + n];
  }
  Y[idx] = act ? gelu_f(acc) : acc;
}

// grouped conv3 (groups=32, 16 in / 16 out per group), epilogue *BN_S then GELU
__global__ void gconv3_kernel(const float* __restrict__ X, const float* __restrict__ w,
                              const float* __restrict__ bsrc, float* __restrict__ Y) {
  int idx = blockIdx.x * blockDim.x + threadIdx.x;
  if (idx >= BL * Nc) return;
  int o = idx & (Nc - 1);
  int l = (idx >> 9) & (Ls - 1);
  int b = idx >> 20;
  int g = o >> 4;
  float acc = bsrc[o];
  for (int t = 0; t < 3; ++t) {
    int lt = l + t - 1;
    if (lt < 0 || lt >= Ls) continue;
    const float* xp = X + ((size_t)(b * Ls + lt)) * Nc + (g << 4);
#pragma unroll
    for (int i = 0; i < 16; ++i) acc += w[(o * 16 + i) * 3 + t] * xp[i];
  }
  Y[idx] = gelu_f(acc * BN_S);
}

// ---------------------------------------------------------------------------
// LayerNorm over L (per (b,n), no affine)  — two passes, coalesced over n
// ---------------------------------------------------------------------------
__global__ void lnL_stats_kernel(const float* __restrict__ X, float* __restrict__ mean,
                                 float* __restrict__ rstd) {
  int idx = blockIdx.x * blockDim.x + threadIdx.x;
  if (idx >= Bb * Nc) return;
  int n = idx & (Nc - 1);
  int b = idx >> 9;
  const float* p = X + (size_t)b * Ls * Nc + n;
  float s = 0.0f, ss = 0.0f;
  for (int l = 0; l < Ls; ++l) { float v = p[(size_t)l * Nc]; s += v; ss += v * v; }
  float m = s * (1.0f / (float)Ls);
  float var = ss * (1.0f / (float)Ls) - m * m;
  mean[idx] = m;
  rstd[idx] = rsqrtf(fmaxf(var, 0.0f) + 1e-5f);
}
__global__ void lnL_apply_kernel(float* __restrict__ X, const float* __restrict__ mean,
                                 const float* __restrict__ rstd) {
  int idx = blockIdx.x * blockDim.x + threadIdx.x;
  if (idx >= BL * Nc) return;
  int sn = ((idx >> 20) << 9) | (idx & (Nc - 1));
  X[idx] = (X[idx] - mean[sn]) * rstd[sn];
}

// ---------------------------------------------------------------------------
// LayerNorm over N (per row, affine, optional residual)
// ---------------------------------------------------------------------------
__global__ __launch_bounds__(256) void lnN_kernel(const float* __restrict__ in,
                                                  const float* __restrict__ res,
                                                  const float* __restrict__ gwt,
                                                  const float* __restrict__ bwt,
                                                  float* __restrict__ out) {
  __shared__ float sv[Nc];
  __shared__ float r1[256], r2[256];
  size_t row = blockIdx.x;
  int t = threadIdx.x;
  float s = 0.0f, ss = 0.0f;
  for (int n = t; n < Nc; n += 256) {
    float v = in[row * Nc + n] + (res ? res[row * Nc + n] : 0.0f);
    sv[n] = v; s += v; ss += v * v;
  }
  r1[t] = s; r2[t] = ss;
  __syncthreads();
  for (int o = 128; o > 0; o >>= 1) {
    if (t < o) { r1[t] += r1[t + o]; r2[t] += r2[t + o]; }
    __syncthreads();
  }
  float m  = r1[0] * (1.0f / (float)Nc);
  float var = r2[0] * (1.0f / (float)Nc) - m * m;
  float rs = rsqrtf(fmaxf(var, 0.0f) + 1e-5f);
  for (int n = t; n < Nc; n += 256) out[row * Nc + n] = (sv[n] - m) * rs * gwt[n] + bwt[n];
}

// ---------------------------------------------------------------------------
// Gate inputs: mean/std over L of original x -> gin (B,1025), gin2 (B,513)
// ---------------------------------------------------------------------------
__global__ void stats_gin_kernel(const float* __restrict__ X, float* __restrict__ gin,
                                 float* __restrict__ gin2) {
  int idx = blockIdx.x * blockDim.x + threadIdx.x;
  if (idx >= Bb * Nc) return;
  int n = idx & (Nc - 1);
  int b = idx >> 9;
  const float* p = X + (size_t)b * Ls * Nc + n;
  float s = 0.0f, ss = 0.0f;
  for (int l = 0; l < Ls; ++l) { float v = p[(size_t)l * Nc]; s += v; ss += v * v; }
  float m = s * (1.0f / (float)Ls);
  float var1 = (ss - (float)Ls * m * m) * (1.0f / (float)(Ls - 1));
  float sd = sqrtf(fmaxf(var1, 0.0f));
  gin[b * 1025 + n]       = m;
  gin[b * 1025 + 512 + n] = sd;
  gin2[b * 513 + n]       = m;
  if (n == 0) { gin[b * 1025 + 1024] = 2.048f; gin2[b * 513 + 512] = 2.048f; }
}

// small gate MLP: F -> 64 (gelu) -> 3 -> softmax  (one block per batch, 64 threads)
__global__ void gatemlp_kernel(const float* __restrict__ gin, int F,
                               const float* __restrict__ w1, const float* __restrict__ b1,
                               const float* __restrict__ w2, const float* __restrict__ b2,
                               float* __restrict__ out) {
  __shared__ float h[64];
  __shared__ float lg[3];
  int b = blockIdx.x;
  int j = threadIdx.x;
  const float* g = gin + (size_t)b * F;
  float acc = b1[j];
  for (int f = 0; f < F; ++f) acc += w1[(size_t)j * F + f] * g[f];
  h[j] = gelu_f(acc);
  __syncthreads();
  if (j < 3) {
    float a = b2[j];
    for (int e = 0; e < 64; ++e) a += w2[j * 64 + e] * h[e];
    lg[j] = a;
  }
  __syncthreads();
  if (j == 0) {
    float mx = fmaxf(lg[0], fmaxf(lg[1], lg[2]));
    float e0 = expf(lg[0] - mx), e1 = expf(lg[1] - mx), e2 = expf(lg[2] - mx);
    float s = e0 + e1 + e2;
    out[b * 3 + 0] = e0 / s; out[b * 3 + 1] = e1 / s; out[b * 3 + 2] = e2 / s;
  }
}

__global__ void wsum3_kernel(const float* __restrict__ A, const float* __restrict__ Bp,
                             const float* __restrict__ Cp, const float* __restrict__ w3,
                             float* __restrict__ out) {
  int idx = blockIdx.x * blockDim.x + threadIdx.x;
  if (idx >= BL * Nc) return;
  int b = idx >> 20;
  out[idx] = w3[b * 3] * A[idx] + w3[b * 3 + 1] * Bp[idx] + w3[b * 3 + 2] * Cp[idx];
}

// ---------------------------------------------------------------------------
// Router: stats over batch -> normed (L,N); sim_s = normed^T @ normed; top-45 mask
// ---------------------------------------------------------------------------
__global__ void statsB_kernel(const float* __restrict__ X2, float* __restrict__ normed) {
  int idx = blockIdx.x * blockDim.x + threadIdx.x;
  if (idx >= Ls * Nc) return;
  int n = idx & (Nc - 1);
  int l = idx >> 9;
  float s = 0.0f, ss = 0.0f;
  for (int b = 0; b < Bb; ++b) {
    float v = X2[((size_t)(b * Ls + l)) * Nc + n];
    s += v; ss += v * v;
  }
  float m = s * (1.0f / (float)Bb);
  float var1 = (ss - (float)Bb * m * m) * (1.0f / (float)(Bb - 1));
  normed[idx] = m / (sqrtf(fmaxf(var1, 0.0f)) + 1e-5f);
}

__global__ __launch_bounds__(512) void sim_kernel(const float* __restrict__ normed,
                                                  float* __restrict__ sim) {
  int m = blockIdx.x;
  int n = threadIdx.x;
  float acc = 0.0f;
  for (int k = 0; k < Ls; ++k)
    acc += normed[(size_t)k * Nc + m] * normed[(size_t)k * Nc + n];
  sim[(size_t)m * Nc + n] = (m == n) ? -1e9f : acc;
}

__global__ void smask_kernel(const float* __restrict__ sim, unsigned char* __restrict__ smask) {
  int n = blockIdx.x * blockDim.x + threadIdx.x;
  if (n >= Nc) return;
  unsigned char* mr = smask + (size_t)n * Nc;
  const float* sr = sim + (size_t)n * Nc;
  for (int m = 0; m < Nc; ++m) mr[m] = 0;
  for (int j = 0; j < STOPK; ++j) {
    float best = -3.4e38f; int bi = 0;
    for (int m = 0; m < Nc; ++m) {
      if (!mr[m]) { float v = sr[m]; if (v > best) { best = v; bi = m; } }
    }
    mr[bi] = 1;
  }
}

// mean / sqrt(var+1e-5) over L of x2 -> tf (B,N,2) + cmean (B,N)
__global__ void stats_tf_kernel(const float* __restrict__ X2, float* __restrict__ tf,
                                float* __restrict__ cmean) {
  int idx = blockIdx.x * blockDim.x + threadIdx.x;
  if (idx >= Bb * Nc) return;
  int n = idx & (Nc - 1);
  int b = idx >> 9;
  const float* p = X2 + (size_t)b * Ls * Nc + n;
  float s = 0.0f, ss = 0.0f;
  for (int l = 0; l < Ls; ++l) { float v = p[(size_t)l * Nc]; s += v; ss += v * v; }
  float m = s * (1.0f / (float)Ls);
  float var1 = (ss - (float)Ls * m * m) * (1.0f / (float)(Ls - 1));
  tf[idx * 2]     = m;
  tf[idx * 2 + 1] = sqrtf(fmaxf(var1, 0.0f) + 1e-5f);
  cmean[idx] = m;
}

// variable features -> Q, K, imp
__global__ void vf_kernel(const float* __restrict__ tf, const float* __restrict__ ve,
                          const float* __restrict__ tpw, const float* __restrict__ tpb,
                          const float* __restrict__ qw, const float* __restrict__ qb,
                          const float* __restrict__ kw, const float* __restrict__ kb,
                          const float* __restrict__ iw, const float* __restrict__ ib,
                          float* __restrict__ Q, float* __restrict__ Kx,
                          float* __restrict__ imp) {
  int idx = blockIdx.x * blockDim.x + threadIdx.x;
  if (idx >= Bb * Nc) return;
  int n = idx & (Nc - 1);
  float f0 = tf[idx * 2], f1 = tf[idx * 2 + 1];
  float vfv[2 * Hh];
#pragma unroll
  for (int h = 0; h < Hh; ++h) vfv[h] = ve[n * Hh + h];
#pragma unroll
  for (int h = 0; h < Hh; ++h) vfv[Hh + h] = tpw[h * 2] * f0 + tpw[h * 2 + 1] * f1 + tpb[h];
  for (int h = 0; h < Hh; ++h) {
    float a = qb[h];
#pragma unroll
    for (int f = 0; f < 2 * Hh; ++f) a += qw[h * 2 * Hh + f] * vfv[f];
    Q[(size_t)idx * Hh + h] = a;
  }
  for (int h = 0; h < Hh; ++h) {
    float a = kb[h];
#pragma unroll
    for (int f = 0; f < 2 * Hh; ++f) a += kw[h * 2 * Hh + f] * vfv[f];
    Kx[(size_t)idx * Hh + h] = a;
  }
  float a = ib[0];
#pragma unroll
  for (int f = 0; f < 2 * Hh; ++f) a += iw[f] * vfv[f];
  imp[idx] = sigm_f(a);
}

// per (b,n): sims over 45 statically-allowed vars, top-27, softmax
__global__ void attn_topk_kernel(const float* __restrict__ Q, const float* __restrict__ Kx,
                                 const unsigned char* __restrict__ smask,
                                 float* __restrict__ aw, int* __restrict__ ti) {
  int idx = blockIdx.x * blockDim.x + threadIdx.x;
  if (idx >= Bb * Nc) return;
  int b = idx >> 9;
  int n = idx & (Nc - 1);
  float q[Hh];
  const float* qp = Q + (size_t)idx * Hh;
#pragma unroll
  for (int h = 0; h < Hh; ++h) q[h] = qp[h];
  float sv[STOPK]; int si[STOPK]; int cnt = 0;
  const unsigned char* mrow = smask + (size_t)n * Nc;
  const float* kbase = Kx + (size_t)b * Nc * Hh;
  for (int m = 0; m < Nc; ++m) {
    if (mrow[m] && cnt < STOPK) {
      const float* kp = kbase + m * Hh;
      float s = 0.0f;
#pragma unroll
      for (int h = 0; h < Hh; ++h) s += q[h] * kp[h];
      sv[cnt] = s * 0.25f;  // 1/sqrt(H)
      si[cnt] = m;
      ++cnt;
    }
  }
  for (int j = 0; j < TOPK; ++j) {
    int best = j;
    for (int k = j + 1; k < cnt; ++k) if (sv[k] > sv[best]) best = k;
    float tv = sv[j]; sv[j] = sv[best]; sv[best] = tv;
    int tq = si[j]; si[j] = si[best]; si[best] = tq;
  }
  float mx = sv[0], ssum = 0.0f;
  float ex[TOPK];
  for (int j = 0; j < TOPK; ++j) { ex[j] = expf(sv[j] - mx); ssum += ex[j]; }
  float inv = 1.0f / ssum;
  float* ap = aw + (size_t)idx * TOPK;
  int* ip = ti + (size_t)idx * TOPK;
  for (int j = 0; j < TOPK; ++j) { ap[j] = ex[j] * inv; ip[j] = si[j]; }
}

// routed + ring combine: block per (b,l), x2 row staged in LDS
__global__ __launch_bounds__(512) void routed_kernel(const float* __restrict__ X2,
                                                     const float* __restrict__ aw,
                                                     const int* __restrict__ ti,
                                                     const float* __restrict__ imp,
                                                     float* __restrict__ out) {
  __shared__ float row[Nc];
  int bl = blockIdx.x;
  int b = bl >> 11;
  int n = threadIdx.x;
  row[n] = X2[(size_t)bl * Nc + n];
  __syncthreads();
  const float* ap = aw + ((size_t)b * Nc + n) * TOPK;
  const int* ip = ti + ((size_t)b * Nc + n) * TOPK;
  float acc = 0.0f;
#pragma unroll
  for (int j = 0; j < TOPK; ++j) acc += ap[j] * row[ip[j]];
  float im = imp[b * Nc + n];
  out[(size_t)bl * Nc + n] = im * acc + (1.0f - im) * row[n];
}

// ---------------------------------------------------------------------------
// Ring-Star centers
// ---------------------------------------------------------------------------
__global__ void cwsm_kernel(const float* __restrict__ vcs, float* __restrict__ cw) {
  int n = blockIdx.x * blockDim.x + threadIdx.x;
  if (n >= Nc) return;
  float mx = -3.4e38f;
  for (int c = 0; c < Cc; ++c) mx = fmaxf(mx, vcs[n * Cc + c]);
  float e[Cc], s = 0.0f;
  for (int c = 0; c < Cc; ++c) { e[c] = expf(vcs[n * Cc + c] - mx); s += e[c]; }
  float inv = 1.0f / s;
  for (int c = 0; c < Cc; ++c) cw[n * Cc + c] = e[c] * inv;
}

__global__ void xct_kernel(const float* __restrict__ X2, const float* __restrict__ cw,
                           float* __restrict__ tb) {
  int idx = blockIdx.x * blockDim.x + threadIdx.x;
  if (idx >= BL * Cc) return;
  int c = idx & (Cc - 1);
  int rowi = idx >> 4;
  const float* xp = X2 + (size_t)rowi * Nc;
  float acc = 0.0f;
  for (int n = 0; n < Nc; ++n) acc += xp[n] * cw[n * Cc + c];
  tb[idx] = acc;
}

__global__ void xc_kernel(const float* __restrict__ tb, const float* __restrict__ vpw,
                          const float* __restrict__ vpb, float* __restrict__ xc) {
  int idx = blockIdx.x * blockDim.x + threadIdx.x;
  if (idx >= BL * Dd) return;
  int d = idx & (Dd - 1);
  int rowi = idx >> 6;
  const float* tp = tb + (size_t)rowi * Cc;
  float acc = vpb[d];
#pragma unroll
  for (int c = 0; c < Cc; ++c) acc += tp[c] * vpw[d * Cc + c];
  xc[idx] = acc;
}

// column sums of cp_w2 (c2v contracts both c and d, so only colsum(W2) is needed)
__global__ void w2s_kernel(const float* __restrict__ w2, const float* __restrict__ b2,
                           float* __restrict__ out) {
  int e = threadIdx.x;  // 64 threads
  float s = 0.0f;
  for (int d = 0; d < Dd; ++d) s += w2[d * Dd + e];
  out[e] = s;
  if (e == 0) {
    float bs = 0.0f;
    for (int d = 0; d < Dd; ++d) bs += b2[d];
    out[64] = bs;
  }
}

// cu_sum[b,l,c] = b2s + sum_j w2s[j]*gelu(hx[b,l,j] + hc[b,c,j] + b1[j])
__global__ void cu_kernel(const float* __restrict__ hx, const float* __restrict__ hc,
                          const float* __restrict__ b1, const float* __restrict__ w2s,
                          float* __restrict__ cu) {
  int idx = blockIdx.x * blockDim.x + threadIdx.x;
  if (idx >= BL * Cc) return;
  int c = idx & (Cc - 1);
  int rowi = idx >> 4;
  int b = rowi >> 11;
  const float* hxp = hx + (size_t)rowi * Dd;
  const float* hcp = hc + ((size_t)((b << 4) + c)) * Dd;
  float acc = w2s[64];
  for (int j = 0; j < Dd; ++j) acc += w2s[j] * gelu_f(hxp[j] + hcp[j] + b1[j]);
  cu[idx] = acc;
}

__global__ void c2v_kernel(const float* __restrict__ cu, const float* __restrict__ cw,
                           float* __restrict__ out) {
  int idx = blockIdx.x * blockDim.x + threadIdx.x;
  if (idx >= BL * Nc) return;
  int n = idx & (Nc - 1);
  int rowi = idx >> 9;
  const float* cp = cu + (size_t)rowi * Cc;
  const float* wp = cw + n * Cc;
  float acc = 0.0f;
#pragma unroll
  for (int c = 0; c < Cc; ++c) acc += wp[c] * cp[c];
  out[idx] = acc;
}

__global__ void mix_kernel(float* __restrict__ ring, const float* __restrict__ c2v,
                           const float* __restrict__ gate) {
  int idx = blockIdx.x * blockDim.x + threadIdx.x;
  if (idx >= BL * Nc) return;
  float g = gate[idx];
  ring[idx] = g * ring[idx] + (1.0f - g) * c2v[idx];
}

// ---------------------------------------------------------------------------
// Host orchestration
// ---------------------------------------------------------------------------
extern "C" void kernel_launch(void* const* d_in, const int* in_sizes, int n_in,
                              void* d_out, int out_size, void* d_ws, size_t ws_size,
                              hipStream_t stream) {
  (void)in_sizes; (void)n_in; (void)out_size; (void)ws_size;
  const float* X          = (const float*)d_in[0];
  const float* t1_w       = (const float*)d_in[1];
  const float* t1_b       = (const float*)d_in[2];
  const float* t2_w1      = (const float*)d_in[3];
  const float* t2_b1      = (const float*)d_in[4];
  const float* t2_w2      = (const float*)d_in[5];
  const float* t2_b2      = (const float*)d_in[6];
  const float* t3_dw1     = (const float*)d_in[7];
  const float* t3_db1     = (const float*)d_in[8];
  const float* t3_pw1     = (const float*)d_in[9];
  const float* t3_pb1     = (const float*)d_in[10];
  const float* t3_dw2     = (const float*)d_in[11];
  const float* t3_db2     = (const float*)d_in[12];
  const float* t3_pw2     = (const float*)d_in[13];
  const float* t3_pb2     = (const float*)d_in[14];
  const float* htm_g_w1   = (const float*)d_in[15];
  const float* htm_g_b1   = (const float*)d_in[16];
  const float* htm_g_w2   = (const float*)d_in[17];
  const float* htm_g_b2   = (const float*)d_in[18];
  const float* htm_proj_w = (const float*)d_in[19];
  const float* htm_proj_b = (const float*)d_in[20];
  const float* htm_ln_g   = (const float*)d_in[21];
  const float* htm_ln_b   = (const float*)d_in[22];
  const float* msc_w3     = (const float*)d_in[23];
  const float* msc_w5     = (const float*)d_in[24];
  const float* msc_w7     = (const float*)d_in[25];
  const float* msc_proj_w = (const float*)d_in[26];
  const float* msc_proj_b = (const float*)d_in[27];
  const float* ht_g_w1    = (const float*)d_in[28];
  const float* ht_g_b1    = (const float*)d_in[29];
  const float* ht_g_w2    = (const float*)d_in[30];
  const float* ht_g_b2    = (const float*)d_in[31];
  const float* ht_proj_w  = (const float*)d_in[32];
  const float* ht_proj_b  = (const float*)d_in[33];
  const float* ht_ln_g    = (const float*)d_in[34];
  const float* ht_ln_b    = (const float*)d_in[35];
  const float* var_embed  = (const float*)d_in[36];
  const float* tp_w       = (const float*)d_in[37];
  const float* tp_b       = (const float*)d_in[38];
  const float* q_w        = (const float*)d_in[39];
  const float* q_b        = (const float*)d_in[40];
  const float* k_w        = (const float*)d_in[41];
  const float* k_b        = (const float*)d_in[42];
  const float* imp_w      = (const float*)d_in[43];
  const float* imp_b      = (const float*)d_in[44];
  const float* cg_w       = (const float*)d_in[45];
  const float* cg_b       = (const float*)d_in[46];
  const float* vcs        = (const float*)d_in[47];
  const float* vp_w       = (const float*)d_in[48];
  const float* vp_b       = (const float*)d_in[49];
  const float* cp_w1      = (const float*)d_in[50];
  const float* cp_b1      = (const float*)d_in[51];
  const float* cp_w2      = (const float*)d_in[52];
  const float* cp_b2      = (const float*)d_in[53];
  const float* gate_w     = (const float*)d_in[54];
  const float* gate_b     = (const float*)d_in[55];
  const float* fus_w      = (const float*)d_in[56];
  const float* fus_b      = (const float*)d_in[57];
  const float* ln_g       = (const float*)d_in[58];
  const float* ln_b       = (const float*)d_in[59];
  float* OUT = (float*)d_out;

  // ---------------- workspace layout (floats) ----------------
  float* Wp = (float*)d_ws;
  const size_t SZB = (size_t)BL * Nc;
  float* G0 = Wp;
  float* G1 = Wp + SZB;
  float* G2 = Wp + 2 * SZB;
  float* G3 = Wp + 3 * SZB;
  size_t off = 4 * SZB;
  auto take = [&](size_t nfl) -> float* {
    float* p = Wp + off;
    off += (nfl + 63) & ~(size_t)63;
    return p;
  };
  float* wt3    = take((size_t)3 * Nc * Nc);
  float* wt5    = take((size_t)5 * Nc * Nc);
  float* wt7    = take((size_t)7 * Nc * Nc);
  float* gin    = take((size_t)Bb * 1025);
  float* gin2   = take((size_t)Bb * 513);
  float* wgt    = take(64);
  float* gwb    = take(64);
  float* lnm    = take((size_t)Bb * Nc);
  float* lnr    = take((size_t)Bb * Nc);
  float* tf     = take((size_t)Bb * Nc * 2);
  float* cmean  = take((size_t)Bb * Nc);
  float* normed = take((size_t)Ls * Nc);
  float* simb   = take((size_t)Nc * Nc);
  unsigned char* smask = (unsigned char*)take((size_t)Nc * Nc / 4);
  float* Qb     = take((size_t)Bb * Nc * Hh);
  float* Kb     = take((size_t)Bb * Nc * Hh);
  float* impb   = take((size_t)Bb * Nc);
  float* awb    = take((size_t)Bb * Nc * TOPK);
  int*   tib    = (int*)take((size_t)Bb * Nc * TOPK);
  float* cwb    = take((size_t)Nc * Cc);
  float* tb     = take((size_t)BL * Cc);
  float* xcb    = take((size_t)BL * Dd);
  float* cenb   = take((size_t)Bb * Cc * Dd);
  float* hxb    = take((size_t)BL * Dd);
  float* hcb    = take((size_t)Bb * Cc * Dd);
  float* cub    = take((size_t)BL * Cc);
  float* w2sb   = take(128);

  auto gemm = [&](const float* A, const float* Wt, const float* bias, float* Y,
                  int M, int K, int Nout, int ldw, int colOff, float alpha, int flags) {
    dim3 grid((Nout + 127) / 128, M / 16);
    gemm_lin_kernel<<<grid, 256, 0, stream>>>(A, Wt, bias, Y, K, Nout, ldw, colOff,
                                              alpha, flags);
  };
  auto gemm_conv = [&](const float* A, const float* Wt, float* Y,
                       int M, int K, int Nout, int taps, int dil, int pad) {
    dim3 grid((Nout + 127) / 128, M / 16);
    gemm_conv_kernel<<<grid, 256, 0, stream>>>(A, Wt, Y, K, Nout, taps, dil, pad, Ls);
  };

  const int gBLN = (BL * Nc + 255) / 256;
  const int gBN  = (Bb * Nc + 255) / 256;

  // --- repack full-conv weights to tap-major ---
  repack_kernel<<<(3 * Nc * Nc + 255) / 256, 256, 0, stream>>>(msc_w3, wt3, Nc, Nc, 3);
  repack_kernel<<<(5 * Nc * Nc + 255) / 256, 256, 0, stream>>>(msc_w5, wt5, Nc, Nc, 5);
  repack_kernel<<<(7 * Nc * Nc + 255) / 256, 256, 0, stream>>>(msc_w7, wt7, Nc, Nc, 7);

  // --- gates (use original x) ---
  stats_gin_kernel<<<gBN, 256, 0, stream>>>(X, gin, gin2);
  gatemlp_kernel<<<Bb, 64, 0, stream>>>(gin, 1025, htm_g_w1, htm_g_b1, htm_g_w2, htm_g_b2, wgt);
  gatemlp_kernel<<<Bb, 64, 0, stream>>>(gin2, 513, ht_g_w1, ht_g_b1, ht_g_w2, ht_g_b2, gwb);

  // --- HTM branch t1 ---
  avgpool7_kernel<<<gBLN, 256, 0, stream>>>(X, G0);
  gemm(G0, t1_w, t1_b, G1, BL, Nc, Nc, Nc, 0, BN_S, GF_BIAS);
  lnL_stats_kernel<<<gBN, 256, 0, stream>>>(G1, lnm, lnr);
  lnL_apply_kernel<<<gBLN, 256, 0, stream>>>(G1, lnm, lnr);
  // --- t2 ---
  gconv3_kernel<<<gBLN, 256, 0, stream>>>(X, t2_w1, t2_b1, G0);
  gemm(G0, t2_w2, t2_b2, G2, BL, Nc, Nc, Nc, 0, 1.0f, GF_BIAS);
  lnL_stats_kernel<<<gBN, 256, 0, stream>>>(G2, lnm, lnr);
  lnL_apply_kernel<<<gBLN, 256, 0, stream>>>(G2, lnm, lnr);
  // --- t3 ---
  dwconv3_kernel<<<gBLN, 256, 0, stream>>>(X, t3_dw1, t3_db1, G0, 1);
  gemm(G0, t3_pw1, t3_pb1, G3, BL, Nc, Nc, Nc, 0, BN_S, GF_BIAS | GF_GELU);
  dwconv3_kernel<<<gBLN, 256, 0, stream>>>(G3, t3_dw2, t3_db2, G0, 0);
  gemm(G0, t3_pw2, t3_pb2, G3, BL, Nc, Nc, Nc, 0, 1.0f, GF_BIAS);
  lnL_stats_kernel<<<gBN, 256, 0, stream>>>(G3, lnm, lnr);
  lnL_apply_kernel<<<gBLN, 256, 0, stream>>>(G3, lnm, lnr);
  // --- fuse + proj + LN(residual x) -> htm in G2 ---
  wsum3_kernel<<<gBLN, 256, 0, stream>>>(G1, G2, G3, wgt, G0);
  gemm(G0, htm_proj_w, htm_proj_b, G1, BL, Nc, Nc, Nc, 0, 1.0f, GF_BIAS);
  lnN_kernel<<<BL, 256, 0, stream>>>(G1, X, htm_ln_g, htm_ln_b, G2);

  // --- MultiScaleConv: three tap GEMMs + split-column projection -> msc in G1 ---
  gemm_conv(X, wt3, G0, BL, Nc, Nc, 3, 1, 1);
  gemm(G0, msc_proj_w, nullptr, G1, BL, Nc, Nc, 1536, 0, 1.0f, 0);
  gemm_conv(X, wt5, G0, BL, Nc, Nc, 5, 2, 4);
  gemm(G0, msc_proj_w, nullptr, G1, BL, Nc, Nc, 1536, 512, 1.0f, GF_ACCUM);
  gemm_conv(X, wt7, G0, BL, Nc, Nc, 7, 3, 9);
  gemm(G0, msc_proj_w, msc_proj_b, G1, BL, Nc, Nc, 1536, 1024, 1.0f,
       GF_ACCUM | GF_BIAS);

  // --- hybrid combine + proj(GELU) + LN -> x2 in G2 ---
  wsum3_kernel<<<gBLN, 256, 0, stream>>>(G2, G1, X, gwb, G0);
  gemm(G0, ht_proj_w, ht_proj_b, G3, BL, Nc, Nc, Nc, 0, 1.0f, GF_BIAS | GF_GELU);
  lnN_kernel<<<BL, 256, 0, stream>>>(G3, nullptr, ht_ln_g, ht_ln_b, G2);

  // --- router static mask ---
  statsB_kernel<<<(Ls * Nc + 255) / 256, 256, 0, stream>>>(G2, normed);
  sim_kernel<<<Nc, Nc, 0, stream>>>(normed, simb);
  smask_kernel<<<2, 256, 0, stream>>>(simb, smask);
  // --- variable features / sparse attention ---
  stats_tf_kernel<<<gBN, 256, 0, stream>>>(G2, tf, cmean);
  vf_kernel<<<gBN, 256, 0, stream>>>(tf, var_embed, tp_w, tp_b, q_w, q_b, k_w, k_b,
                                     imp_w, imp_b, Qb, Kb, impb);
  attn_topk_kernel<<<gBN, 256, 0, stream>>>(Qb, Kb, smask, awb, tib);
  routed_kernel<<<BL, 512, 0, stream>>>(G2, awb, tib, impb, G0);  // ring -> G0

  // --- ring-star centers ---
  cwsm_kernel<<<2, 256, 0, stream>>>(vcs, cwb);
  xct_kernel<<<(BL * Cc + 255) / 256, 256, 0, stream>>>(G2, cwb, tb);
  xc_kernel<<<(BL * Dd + 255) / 256, 256, 0, stream>>>(tb, vp_w, vp_b, xcb);
  gemm(cmean, cg_w, cg_b, cenb, Bb, Nc, Cc * Dd, Nc, 0, 1.0f, GF_BIAS);
  gemm(xcb, cp_w1, nullptr, hxb, BL, Dd, Dd, Dd, 0, 1.0f, 0);
  gemm(cenb, cp_w1, nullptr, hcb, Bb * Cc, Dd, Dd, Dd, 0, 1.0f, 0);
  w2s_kernel<<<1, 64, 0, stream>>>(cp_w2, cp_b2, w2sb);
  cu_kernel<<<(BL * Cc + 255) / 256, 256, 0, stream>>>(hxb, hcb, cp_b1, w2sb, cub);
  c2v_kernel<<<gBLN, 256, 0, stream>>>(cub, cwb, G1);  // c2v -> G1

  // --- gate (2N-input split GEMM), mix, fuse, output LN ---
  gemm(G0, gate_w, nullptr, G3, BL, Nc, Nc, 1024, 0, 1.0f, 0);
  gemm(G1, gate_w, gate_b, G3, BL, Nc, Nc, 1024, 512, 1.0f,
       GF_ACCUM | GF_BIAS | GF_SIGM);
  mix_kernel<<<gBLN, 256, 0, stream>>>(G0, G1, G3);
  gemm(G0, fus_w, fus_b, G3, BL, Nc, Nc, Nc, 0, 1.0f, GF_BIAS);
  lnN_kernel<<<BL, 256, 0, stream>>>(G3, G2, ln_g, ln_b, OUT);
}
